// QuaternionLinear_15384572854307
// MI455X (gfx1250) — matrix operand order repaired and checked by reading, not compiled, not run on hardware
//
#include <hip/hip_runtime.h>
#include <hip/hip_bf16.h>
#include <stdint.h>

// ---------------------------------------------------------------------------
// QuaternionLinear on MI455X (gfx1250): y = x @ W_hamilton + b_fused
//   x:   [8192, 4096] f32
//   W*:  4x [1024, 1024] f32  (fused into 4096x4096 with Hamilton signs)
//   y:   [8192, 4096] f32
// Compute-bound GEMM -> bf16 WMMA (v_wmma_f32_16x16x32_bf16), f32 accumulate.
// Tile staging: async global->LDS DMA (ASYNCcnt) when the builtins exist,
// else register-pipelined global_load_b128 + ds_store_b128.
// ---------------------------------------------------------------------------

typedef __attribute__((ext_vector_type(16))) __bf16 bf16x16;
typedef __attribute__((ext_vector_type(8)))  float  floatx8;

union AFrag {
    bf16x16 v;
    uint4   q[2];
};

#define SZ_B   8192
#define SZ_K   4096
#define SZ_N   4096
#define SZ_Q   1024

#define BM 128
#define BN 128
#define BKK 64
#define LDK (BKK + 8)   // +8 shorts (16B) pad: LDS row stride 144B, kills bank conflicts

#if __has_builtin(__builtin_amdgcn_global_load_async_to_lds_b128) && \
    __has_builtin(__builtin_amdgcn_s_wait_asynccnt)
#define QL_ASYNC 1
#else
#define QL_ASYNC 0
#endif

__device__ __forceinline__ unsigned short f2bf(float f) {
    union { __bf16 h; unsigned short s; } u;
    u.h = (__bf16)f;
    return u.s;
}

// ---------------- Prologue 1: x f32 -> bf16 (8 elems / thread) -------------
__global__ __launch_bounds__(256) void cvt_x_kernel(const float* __restrict__ x,
                                                    unsigned short* __restrict__ xb) {
    size_t i = (size_t)blockIdx.x * 256 + threadIdx.x;   // handles 8 floats
    const float4* xv = (const float4*)x;
    float4 a = xv[i * 2 + 0];
    float4 b = xv[i * 2 + 1];
    uint4 o;
    o.x = (unsigned)f2bf(a.x) | ((unsigned)f2bf(a.y) << 16);
    o.y = (unsigned)f2bf(a.z) | ((unsigned)f2bf(a.w) << 16);
    o.z = (unsigned)f2bf(b.x) | ((unsigned)f2bf(b.y) << 16);
    o.w = (unsigned)f2bf(b.z) | ((unsigned)f2bf(b.w) << 16);
    ((uint4*)xb)[i] = o;
}

// ---------------- Prologue 2: fused, signed, TRANSPOSED weight -------------
// Wt[n][k] (bf16, N-major) = sign(kc,nc) * Wcomp(kc,nc)[k%1024][n%1024]
// Each 32x32 tile lies entirely inside one (kc,nc) quaternion block.
__global__ __launch_bounds__(256) void build_wt_kernel(const float* __restrict__ Wr,
                                                       const float* __restrict__ Wi,
                                                       const float* __restrict__ Wj,
                                                       const float* __restrict__ Wk,
                                                       unsigned short* __restrict__ wt) {
    __shared__ float tile[32][33];
    const int k0 = blockIdx.x * 32;   // input (K) dim of fused matrix
    const int n0 = blockIdx.y * 32;   // output (N) dim of fused matrix
    const int kc = k0 >> 10, nc = n0 >> 10;

    // Hamilton block structure: rows = input chunk kc, cols = output chunk nc
    const int   comp[4][4] = {{0,1,2,3},{1,0,3,2},{2,3,0,1},{3,2,1,0}};
    const float sgn [4][4] = {{ 1.f, 1.f, 1.f, 1.f},
                              {-1.f, 1.f, 1.f,-1.f},
                              {-1.f,-1.f, 1.f, 1.f},
                              {-1.f, 1.f,-1.f, 1.f}};
    const float* Wc[4] = {Wr, Wi, Wj, Wk};
    const float* src = Wc[comp[kc][nc]];
    const float  s   = sgn[kc][nc];

    const int tx = threadIdx.x, ty = threadIdx.y;   // blockDim = (32, 8)
    #pragma unroll
    for (int i = 0; i < 4; ++i) {
        int k = k0 + ty + i * 8;
        int n = n0 + tx;                            // coalesced read along n'
        tile[ty + i * 8][tx] = s * src[(size_t)(k & 1023) * SZ_Q + (n & 1023)];
    }
    __syncthreads();
    #pragma unroll
    for (int i = 0; i < 4; ++i) {
        int n = n0 + ty + i * 8;
        int k = k0 + tx;                            // coalesced write along k
        wt[(size_t)n * SZ_K + k] = f2bf(tile[tx][ty + i * 8]);
    }
}

// ---------------- Prologue 3: fused bias -----------------------------------
__global__ __launch_bounds__(256) void build_bias_kernel(const float* __restrict__ br,
                                                         const float* __restrict__ bi,
                                                         const float* __restrict__ bj,
                                                         const float* __restrict__ bk,
                                                         float* __restrict__ bias) {
    int n = blockIdx.x * 256 + threadIdx.x;
    if (n >= SZ_N) return;
    int nc = n >> 10, j = n & 1023;
    const float S[4][4] = {{1.f,-1.f,-1.f,-1.f},
                           {1.f, 1.f, 1.f,-1.f},
                           {1.f,-1.f, 1.f, 1.f},
                           {1.f, 1.f,-1.f, 1.f}};
    bias[n] = S[nc][0]*br[j] + S[nc][1]*bi[j] + S[nc][2]*bj[j] + S[nc][3]*bk[j];
}

// ---- shared compute stage: 2x(16x16x32) k-steps over one 64-wide LDS slab --
__device__ __forceinline__ void compute_stage(const unsigned short (*Asb)[LDK],
                                              const unsigned short (*Bsb)[LDK],
                                              floatx8 acc[2][4],
                                              int waveM, int waveN, int hi16, int lm) {
    #pragma unroll
    for (int kk = 0; kk < BKK; kk += 32) {
        AFrag a[2], b[4];
        // A 16x32 bf16 frag: lanes 0-15 hold K {0..7,16..23}, lanes 16-31 {8..15,24..31}
        #pragma unroll
        for (int mi = 0; mi < 2; ++mi) {
            const int row = waveM * 32 + mi * 16 + lm;
            const int c0  = kk + (hi16 ? 8  : 0);
            const int c1  = kk + (hi16 ? 24 : 16);
            a[mi].q[0] = *(const uint4*)&Asb[row][c0];
            a[mi].q[1] = *(const uint4*)&Asb[row][c1];
        }
        // B 32x16 bf16 frag: lane=col, lanes 0-15 hold K {0..7,16..23}, lanes 16-31 {8..15,24..31}
        #pragma unroll
        for (int ni = 0; ni < 4; ++ni) {
            const int row = waveN * 64 + ni * 16 + lm;
            const int c0  = kk + (hi16 ? 8 : 0);
            b[ni].q[0] = *(const uint4*)&Bsb[row][c0];
            b[ni].q[1] = *(const uint4*)&Bsb[row][c0 + 16];
        }
        #pragma unroll
        for (int mi = 0; mi < 2; ++mi)
            #pragma unroll
            for (int ni = 0; ni < 4; ++ni)
                acc[mi][ni] = __builtin_amdgcn_wmma_f32_16x16x32_bf16(
                    false, a[mi].v, false, b[ni].v,
                    (short)0, acc[mi][ni], false, false);
    }
}

// ---------------- Main GEMM: 128x128x64 tile, 8 waves (4Mx2N) --------------
// Wave owns 32x64 -> 2x4 accumulators of v_wmma_f32_16x16x32_bf16.
#if QL_ASYNC

// builtin prototype (from diagnostic): pointee is a GCC-style int 4-vector,
// src in AS1 (__device__), dst in AS3 (__shared__)
typedef int v4i_vs __attribute__((vector_size(16)));
typedef __attribute__((address_space(1))) v4i_vs* gptr_b128;
typedef __attribute__((address_space(3))) v4i_vs* lptr_b128;

__device__ __forceinline__ void async_b128(const void* g, void* l) {
    __builtin_amdgcn_global_load_async_to_lds_b128((gptr_b128)(g), (lptr_b128)(l), 0, 0);
}

__global__ __launch_bounds__(256) void qgemm_kernel(const unsigned short* __restrict__ xb,
                                                    const unsigned short* __restrict__ wt,
                                                    const float* __restrict__ bias,
                                                    float* __restrict__ out) {
    __shared__ __align__(16) unsigned short As[2][BM][LDK];
    __shared__ __align__(16) unsigned short Bs[2][BN][LDK];

    const int tid   = threadIdx.x;
    const int wave  = tid >> 5;
    const int lane  = tid & 31;
    const int waveM = wave >> 1;          // 0..3
    const int waveN = wave & 1;           // 0..1
    const int hi16  = lane >> 4;          // lane half (ISA 16-bit frag layout)
    const int lm    = lane & 15;

    const int m0 = blockIdx.y * BM;
    const int n0 = blockIdx.x * BN;

    // cooperative tile DMA mapping: 256 threads, 16-elem (32B) chunks
    const int lr = tid >> 2;              // row 0..63 (and +64)
    const int lc = (tid & 3) * 16;        // chunk within 64-wide K slab

    floatx8 acc[2][4];
    #pragma unroll
    for (int mi = 0; mi < 2; ++mi)
        #pragma unroll
        for (int ni = 0; ni < 4; ++ni)
            #pragma unroll
            for (int e = 0; e < 8; ++e) acc[mi][ni][e] = 0.0f;

    // issue 4 async b128 DMAs (A rows lr, lr+64; B rows lr, lr+64) for stage kt
    #define QL_ISSUE(buf, kt)                                                       \
        do {                                                                        \
            const size_t ka = (size_t)(kt) * BKK + lc;                              \
            async_b128(&xb[(size_t)(m0 + lr)      * SZ_K + ka], &As[buf][lr     ][lc]); \
            async_b128(&xb[(size_t)(m0 + lr + 64) * SZ_K + ka], &As[buf][lr + 64][lc]); \
            async_b128(&wt[(size_t)(n0 + lr)      * SZ_K + ka], &Bs[buf][lr     ][lc]); \
            async_b128(&wt[(size_t)(n0 + lr + 64) * SZ_K + ka], &Bs[buf][lr + 64][lc]); \
        } while (0)

    QL_ISSUE(0, 0);

    const int NK = SZ_K / BKK;            // 64 k-tiles
    for (int kt = 0; kt < NK; ++kt) {
        const int cur = kt & 1;
        if (kt + 1 < NK) {
            QL_ISSUE(cur ^ 1, kt + 1);    // DMA next stage into other buffer
            __builtin_amdgcn_s_wait_asynccnt(4);  // stage-kt DMAs complete
        } else {
            __builtin_amdgcn_s_wait_asynccnt(0);
        }
        if (kt + 2 < NK) {                // L2 prefetch hint two tiles ahead
            const size_t kp = (size_t)(kt + 2) * BKK + lc;
            __builtin_prefetch(&xb[(size_t)(m0 + lr) * SZ_K + kp], 0, 1);
            __builtin_prefetch(&wt[(size_t)(n0 + lr) * SZ_K + kp], 0, 1);
        }
        __syncthreads();                  // all waves' stage-kt data visible
        compute_stage(As[cur], Bs[cur], acc, waveM, waveN, hi16, lm);
        __syncthreads();                  // done reading before next overwrite
    }
    #undef QL_ISSUE

    // ---- epilogue: C/D layout — VGPR v: lanes 0-15 -> M=v, lanes 16-31 -> M=v+8
    float bv[4];
    #pragma unroll
    for (int ni = 0; ni < 4; ++ni)
        bv[ni] = bias[n0 + waveN * 64 + ni * 16 + lm];

    #pragma unroll
    for (int mi = 0; mi < 2; ++mi) {
        #pragma unroll
        for (int ni = 0; ni < 4; ++ni) {
            const int gcol = n0 + waveN * 64 + ni * 16 + lm;
            #pragma unroll
            for (int v = 0; v < 8; ++v) {
                const int grow = m0 + waveM * 32 + mi * 16 + hi16 * 8 + v;
                out[(size_t)grow * SZ_N + gcol] = acc[mi][ni][v] + bv[ni];
            }
        }
    }
}

#else  // !QL_ASYNC — register-pipelined fallback

__global__ __launch_bounds__(256) void qgemm_kernel(const unsigned short* __restrict__ xb,
                                                    const unsigned short* __restrict__ wt,
                                                    const float* __restrict__ bias,
                                                    float* __restrict__ out) {
    __shared__ __align__(16) unsigned short As[BM][LDK];
    __shared__ __align__(16) unsigned short Bs[BN][LDK];

    const int tid   = threadIdx.x;
    const int wave  = tid >> 5;
    const int lane  = tid & 31;
    const int waveM = wave >> 1;
    const int waveN = wave & 1;
    const int hi16  = lane >> 4;
    const int lm    = lane & 15;

    const int m0 = blockIdx.y * BM;
    const int n0 = blockIdx.x * BN;

    const int lr = tid >> 2;
    const int lc = (tid & 3) * 16;

    floatx8 acc[2][4];
    #pragma unroll
    for (int mi = 0; mi < 2; ++mi)
        #pragma unroll
        for (int ni = 0; ni < 4; ++ni)
            #pragma unroll
            for (int e = 0; e < 8; ++e) acc[mi][ni][e] = 0.0f;

    uint4 ra0, ra1, rb0, rb1;
    {
        const size_t ka = (size_t)lc;
        ra0 = *(const uint4*)&xb[(size_t)(m0 + lr)      * SZ_K + ka];
        ra1 = *(const uint4*)&xb[(size_t)(m0 + lr + 64) * SZ_K + ka];
        rb0 = *(const uint4*)&wt[(size_t)(n0 + lr)      * SZ_K + ka];
        rb1 = *(const uint4*)&wt[(size_t)(n0 + lr + 64) * SZ_K + ka];
        *(uint4*)&As[lr     ][lc] = ra0;
        *(uint4*)&As[lr + 64][lc] = ra1;
        *(uint4*)&Bs[lr     ][lc] = rb0;
        *(uint4*)&Bs[lr + 64][lc] = rb1;
    }
    __syncthreads();

    const int NK = SZ_K / BKK;
    for (int kt = 0; kt < NK; ++kt) {
        if (kt + 1 < NK) {
            const size_t ka = (size_t)(kt + 1) * BKK + lc;
            ra0 = *(const uint4*)&xb[(size_t)(m0 + lr)      * SZ_K + ka];
            ra1 = *(const uint4*)&xb[(size_t)(m0 + lr + 64) * SZ_K + ka];
            rb0 = *(const uint4*)&wt[(size_t)(n0 + lr)      * SZ_K + ka];
            rb1 = *(const uint4*)&wt[(size_t)(n0 + lr + 64) * SZ_K + ka];
        }
        if (kt + 2 < NK) {
            const size_t kp = (size_t)(kt + 2) * BKK + lc;
            __builtin_prefetch(&xb[(size_t)(m0 + lr) * SZ_K + kp], 0, 1);
            __builtin_prefetch(&wt[(size_t)(n0 + lr) * SZ_K + kp], 0, 1);
        }
        compute_stage(As, Bs, acc, waveM, waveN, hi16, lm);
        __syncthreads();
        if (kt + 1 < NK) {
            *(uint4*)&As[lr     ][lc] = ra0;
            *(uint4*)&As[lr + 64][lc] = ra1;
            *(uint4*)&Bs[lr     ][lc] = rb0;
            *(uint4*)&Bs[lr + 64][lc] = rb1;
            __syncthreads();
        }
    }

    float bv[4];
    #pragma unroll
    for (int ni = 0; ni < 4; ++ni)
        bv[ni] = bias[n0 + waveN * 64 + ni * 16 + lm];

    #pragma unroll
    for (int mi = 0; mi < 2; ++mi) {
        #pragma unroll
        for (int ni = 0; ni < 4; ++ni) {
            const int gcol = n0 + waveN * 64 + ni * 16 + lm;
            #pragma unroll
            for (int v = 0; v < 8; ++v) {
                const int grow = m0 + waveM * 32 + mi * 16 + hi16 * 8 + v;
                out[(size_t)grow * SZ_N + gcol] = acc[mi][ni][v] + bv[ni];
            }
        }
    }
}

#endif // QL_ASYNC

// ---------------------------------------------------------------------------
extern "C" void kernel_launch(void* const* d_in, const int* in_sizes, int n_in,
                              void* d_out, int out_size, void* d_ws, size_t ws_size,
                              hipStream_t stream) {
    const float* x  = (const float*)d_in[0];
    const float* Wr = (const float*)d_in[1];
    const float* Wi = (const float*)d_in[2];
    const float* Wj = (const float*)d_in[3];
    const float* Wk = (const float*)d_in[4];
    const float* br = (const float*)d_in[5];
    const float* bi = (const float*)d_in[6];
    const float* bj = (const float*)d_in[7];
    const float* bk = (const float*)d_in[8];
    float* out = (float*)d_out;

    // workspace layout
    char* ws = (char*)d_ws;
    const size_t XB_BYTES = (size_t)SZ_B * SZ_K * sizeof(unsigned short); // 67.1 MB
    const size_t WT_BYTES = (size_t)SZ_N * SZ_K * sizeof(unsigned short); // 33.6 MB
    unsigned short* xb   = (unsigned short*)(ws);
    unsigned short* wt   = (unsigned short*)(ws + XB_BYTES);
    float*          bias = (float*)(ws + XB_BYTES + WT_BYTES);

    // 1) x -> bf16
    {
        const size_t nthreads = (size_t)SZ_B * SZ_K / 8;   // 8 floats / thread
        cvt_x_kernel<<<dim3((unsigned)(nthreads / 256)), dim3(256), 0, stream>>>(x, xb);
    }
    // 2) fused transposed bf16 weight
    {
        dim3 grid(SZ_K / 32, SZ_N / 32);   // (128, 128)
        dim3 block(32, 8);
        build_wt_kernel<<<grid, block, 0, stream>>>(Wr, Wi, Wj, Wk, wt);
    }
    // 3) fused bias
    build_bias_kernel<<<dim3(SZ_N / 256), dim3(256), 0, stream>>>(br, bi, bj, bk, bias);

    // 4) WMMA GEMM
    {
        dim3 grid(SZ_N / BN, SZ_B / BM);   // (32, 64)
        qgemm_kernel<<<grid, dim3(256), 0, stream>>>(xb, wt, bias, out);
    }
}
